// ColorDeformConv2d_74887049773277
// MI455X (gfx1250) — matrix operand order repaired and checked by reading, not compiled, not run on hardware
//
#include <hip/hip_runtime.h>
#include <math.h>

typedef unsigned short u16t;
typedef unsigned int   u32t;
typedef __attribute__((ext_vector_type(16))) __bf16        v16bf;
typedef __attribute__((ext_vector_type(8)))  float         v8f;
typedef __attribute__((ext_vector_type(4)))  unsigned int  v4u;

#define BN 4
#define CC 64
#define HH 128
#define WW 128
#define HWSZ (HH * WW)      // 16384
#define KC9 576             // C * 9
#define PSTRIDE 584         // 576 + 8 pad (keeps 16B row alignment)

// Prepped-weight element counts (WMMA A-operand order: [mt][kc][lane][16])
#define WCD_ELEMS   (4 * 4 * 512)     // 8192
#define WOM_ELEMS   (2 * 18 * 512)    // 18432
#define WC_ELEMS    (36 * 18 * 512)   // 331776
#define WCONV_ELEMS (4 * 18 * 512)    // 36864
#define FUSED_ELEMS (BN * CC * HWSZ)  // 4194304

union BFV {
    v16bf v;
    u16t  s[16];
    v4u   q[2];
};

__device__ __forceinline__ u16t f2bf(float f) {
    u32t u = __float_as_uint(f);
    u32t r = u + 0x7FFFu + ((u >> 16) & 1u);   // round-to-nearest-even bf16
    return (u16t)(r >> 16);
}

__device__ __forceinline__ v8f zero8() {
    v8f z;
#pragma unroll
    for (int i = 0; i < 8; ++i) z[i] = 0.0f;
    return z;
}

// A operand from PREPPED bf16 weights: 32 bytes contiguous per lane, zero VALU.
__device__ __forceinline__ v16bf load_a_pre(const u16t* opbase, int mt, int kchunks,
                                            int kc, int lane) {
    BFV u;
    const v4u* p = (const v4u*)(opbase + (((size_t)mt * kchunks + kc) << 9) + (lane << 4));
    u.q[0] = p[0];
    u.q[1] = p[1];
    return u.v;
}

// B operand (32x16 bf16): lane n=lane&15 holds column n; lanes 0-15 K=kc..kc+15,
// lanes 16-31 K=kc+16..kc+31. 16 contiguous bf16 from an LDS row.
__device__ __forceinline__ v16bf load_b_lds(const u16t* rowp, int kc, int lane) {
    BFV u;
    const v4u* p = (const v4u*)(rowp + kc + ((lane & 16) ? 16 : 0));
    u.q[0] = p[0];
    u.q[1] = p[1];
    return u.v;
}

#define WMMA_BF16(A, B, C) \
    __builtin_amdgcn_wmma_f32_16x16x32_bf16(false, (A), false, (B), (short)0, (C), false, false)

// ---------------------------------------------------------------------------
// Weight prep: fp32 row-major [M x Kstride] -> bf16 WMMA A-operand tile order.
// ---------------------------------------------------------------------------
__device__ __forceinline__ void prep_one(const float* src, u16t* dst, int e,
                                         int kchunks, int kstride, int mvalid) {
    const int tile = e >> 9;
    const int within = e & 511;
    const int lane = within >> 4;
    const int i = within & 15;
    const int mt = tile / kchunks;
    const int kc = tile - mt * kchunks;
    const int row = mt * 16 + (lane & 15);
    const int koff = ((lane & 16) ? 16 : 0) + i;
    const int k = kc * 32 + koff;
    float v = 0.0f;
    if (row < mvalid) v = src[(size_t)row * kstride + k];
    dst[e] = f2bf(v);
}

__global__ __launch_bounds__(256) void prep_weights(
    const float* __restrict__ w_cd, const float* __restrict__ w_p,
    const float* __restrict__ w_m, const float* __restrict__ w_c,
    const float* __restrict__ w_conv,
    u16t* __restrict__ wcd_op, u16t* __restrict__ wom_op,
    u16t* __restrict__ wc_op, u16t* __restrict__ wconv_op) {
    const int gid = blockIdx.x * 256 + threadIdx.x;

    if (gid < WCD_ELEMS)   prep_one(w_cd, wcd_op, gid, 4, 128, 64);
    if (gid < WC_ELEMS)    prep_one(w_c, wc_op, gid, 18, KC9, KC9);
    if (gid < WCONV_ELEMS) prep_one(w_conv, wconv_op, gid, 18, KC9, 64);

    if (gid < WOM_ELEMS) {  // composite: rows 0-17 w_p, 18-26 w_m, 27-31 zero
        const int e = gid;
        const int tile = e >> 9;
        const int within = e & 511;
        const int lane = within >> 4;
        const int i = within & 15;
        const int mt = tile / 18;
        const int kc = tile - mt * 18;
        const int row = mt * 16 + (lane & 15);
        const int koff = ((lane & 16) ? 16 : 0) + i;
        const int k = kc * 32 + koff;
        float v = 0.0f;
        if (row < 18)      v = w_p[(size_t)row * KC9 + k];
        else if (row < 27) v = w_m[(size_t)(row - 18) * KC9 + k];
        wom_op[e] = f2bf(v);
    }
}

// ---------------------------------------------------------------------------
// Kernel 1: fused = W_cd[64x128] @ concat(x, ref) + b_cd, bf16 output.
// ---------------------------------------------------------------------------
__global__ __launch_bounds__(128) void stage1_fused_gemm(
    const float* __restrict__ x, const float* __restrict__ ref,
    const u16t* __restrict__ wcd_op, const float* __restrict__ b_cd,
    u16t* __restrict__ fused) {
    __shared__ __align__(16) u16t lds[4][16][136];

    const int tid  = threadIdx.x;
    const int wv   = tid >> 5;
    const int lane = tid & 31;
    const int tile = blockIdx.x * 4 + wv;
    const int pix0 = tile * 16;
    const int b    = pix0 >> 14;
    const int rem  = pix0 & 16383;
    const int h    = rem >> 7;
    const int w0   = rem & 127;

    for (int i = lane; i < 16 * 128; i += 32) {
        const int j = i & 15;
        const int k = i >> 4;
        const float* src = (k < 64) ? x : ref;
        const float v = src[(((size_t)b * 64 + (k & 63)) * 128 + h) * 128 + w0 + j];
        lds[wv][j][k] = f2bf(v);
    }
    __syncthreads();

    v8f acc[4];
#pragma unroll
    for (int mt = 0; mt < 4; ++mt) acc[mt] = zero8();

#pragma unroll
    for (int kc = 0; kc < 4; ++kc) {
        const v16bf bop = load_b_lds(&lds[wv][lane & 15][0], kc * 32, lane);
#pragma unroll
        for (int mt = 0; mt < 4; ++mt) {
            const v16bf aop = load_a_pre(wcd_op, mt, 4, kc, lane);
            acc[mt] = WMMA_BF16(aop, bop, acc[mt]);
        }
    }

    const int hi = (lane >> 4) & 1;
    const int n  = lane & 15;
#pragma unroll
    for (int mt = 0; mt < 4; ++mt) {
#pragma unroll
        for (int r = 0; r < 8; ++r) {
            const int m = mt * 16 + r + hi * 8;
            const float v = acc[mt][r] + b_cd[m];
            fused[(((size_t)b * 64 + m) * 128 + h) * 128 + w0 + n] = f2bf(v);
        }
    }
}

// ---------------------------------------------------------------------------
// Kernel 2: fully fused tail. 32-pixel tiles, 9 waves (288 threads).
//   waves 0-3: off/mod GEMM (2 M x 2 N tiles)
//   all 9 waves: col GEMM (wave w owns M-tiles 4w..4w+3, both N-tiles)
//   waves 0-7: output GEMM (4 M x 2 N tiles)
// ---------------------------------------------------------------------------
__global__ __launch_bounds__(288) void stage2_deform_main(
    const float* __restrict__ x, const u16t* __restrict__ fused,
    const u16t* __restrict__ wom_op, const float* __restrict__ b_p,
    const float* __restrict__ b_m,
    const u16t* __restrict__ wc_op, const float* __restrict__ b_c,
    const u16t* __restrict__ wconv_op, float* __restrict__ out) {
    __shared__ __align__(16) u16t patch[32][PSTRIDE];  // im2col(fused)  [pixel][ck]
    __shared__ __align__(16) u16t vmat[32][PSTRIDE];   // (col+pos)*mod  [pixel][ck]
    __shared__ float omv[32][32];                      // rows 0-17 off, 18-26 sigmoid(mod)
    __shared__ int   gidx[9][32][4];
    __shared__ float gwt[9][32][4];

    const int tid  = threadIdx.x;
    const int wv   = tid >> 5;
    const int lane = tid & 31;
    const int pix0 = blockIdx.x * 32;
    const int b    = pix0 >> 14;
    const int rem  = pix0 & 16383;
    const int h    = rem >> 7;
    const int w0   = rem & 127;

    const int n  = lane & 15;
    const int hi = (lane >> 4) & 1;

    // --- 1. im2col patch: patch[j][c*9+t] = fused[b,c,h+tx-1,w0+j+ty-1]
    for (int i = tid; i < 32 * KC9; i += 288) {
        const int j  = i & 31;
        const int kk = i >> 5;
        const int c  = kk / 9;
        const int t  = kk - c * 9;
        const int hh = h + t / 3 - 1;
        const int ww = w0 + j + (t % 3) - 1;
        u16t v = 0;
        if (hh >= 0 && hh < 128 && ww >= 0 && ww < 128)
            v = fused[(((size_t)b * 64 + c) * 128 + hh) * 128 + ww];
        patch[j][kk] = v;
    }
    __syncthreads();

    // --- 2. off/mod GEMM: M=32 (rows 0-17 off, 18-26 mod), K=576, N=32
    if (wv < 4) {
        const int mt = wv >> 1;
        const int nt = wv & 1;
        v8f acc = zero8();
        for (int kc = 0; kc < 18; ++kc) {
            const v16bf bop = load_b_lds(&patch[nt * 16 + n][0], kc * 32, lane);
            const v16bf aop = load_a_pre(wom_op, mt, 18, kc, lane);
            acc = WMMA_BF16(aop, bop, acc);
        }
#pragma unroll
        for (int r = 0; r < 8; ++r) {
            const int mo = mt * 16 + r + hi * 8;
            if (mo < 27) {
                const float bias = (mo < 18) ? b_p[mo] : b_m[mo - 18];
                const float v = acc[r] + bias;
                omv[mo][nt * 16 + n] = (mo < 18) ? v : (1.0f / (1.0f + __expf(-v)));
            }
        }
    }
    __syncthreads();

    // --- 3. bilinear gather idx/weights: exactly one (tap, pixel) per thread
    {
        const int t  = tid / 32;          // 0..8
        const int nn = tid - t * 32;      // 0..31
        const float ox = omv[t][nn];
        const float oy = omv[9 + t][nn];
        const float cx = (float)(t / 3 - 1);
        const float cy = (float)(t % 3 - 1);
        const float px = (float)(h + 1) + ox + cx;
        const float py = (float)(w0 + nn + 1) + oy + cy;
        const float fx = floorf(px), fy = floorf(py);
        const float q0x = fminf(fmaxf(fx, 0.f), 129.f);
        const float q0y = fminf(fmaxf(fy, 0.f), 129.f);
        const float q1x = fminf(fmaxf(fx + 1.f, 0.f), 129.f);
        const float q1y = fminf(fmaxf(fy + 1.f, 0.f), 129.f);
        const float sx = fminf(fmaxf(px, 0.f), 129.f);
        const float sy = fminf(fmaxf(py, 0.f), 129.f);
        const float glt = (1.f + (q0x - sx)) * (1.f + (q0y - sy));
        const float grb = (1.f - (q1x - sx)) * (1.f - (q1y - sy));
        const float glb = (1.f + (q0x - sx)) * (1.f - (q1y - sy));
        const float grt = (1.f - (q1x - sx)) * (1.f + (q0y - sy));
        const int i0x = (int)q0x - 1, i0y = (int)q0y - 1;
        const int i1x = (int)q1x - 1, i1y = (int)q1y - 1;
        const bool v0x = (i0x >= 0) && (i0x < 128), v0y = (i0y >= 0) && (i0y < 128);
        const bool v1x = (i1x >= 0) && (i1x < 128), v1y = (i1y >= 0) && (i1y < 128);
        gidx[t][nn][0] = (v0x && v0y) ? (i0x * 128 + i0y) : -1;
        gidx[t][nn][1] = (v1x && v1y) ? (i1x * 128 + i1y) : -1;
        gidx[t][nn][2] = (v0x && v1y) ? (i0x * 128 + i1y) : -1;
        gidx[t][nn][3] = (v1x && v0y) ? (i1x * 128 + i0y) : -1;
        gwt[t][nn][0] = glt; gwt[t][nn][1] = grb;
        gwt[t][nn][2] = glb; gwt[t][nn][3] = grt;
    }
    __syncthreads();

    // --- 4. col GEMM (M=576: wave owns M-tiles 4*wv..4*wv+3, N-tiles 0,1)
    //        + tanh/gather/mod epilogue -> vmat (bf16)
    {
        v8f acc[4][2];
#pragma unroll
        for (int q = 0; q < 4; ++q) {
            acc[q][0] = zero8();
            acc[q][1] = zero8();
        }
        for (int kc = 0; kc < 18; ++kc) {
            const v16bf bop0 = load_b_lds(&patch[n][0], kc * 32, lane);
            const v16bf bop1 = load_b_lds(&patch[16 + n][0], kc * 32, lane);
#pragma unroll
            for (int q = 0; q < 4; ++q) {
                const v16bf aop = load_a_pre(wc_op, wv * 4 + q, 18, kc, lane);
                acc[q][0] = WMMA_BF16(aop, bop0, acc[q][0]);
                acc[q][1] = WMMA_BF16(aop, bop1, acc[q][1]);
            }
        }
#pragma unroll
        for (int q = 0; q < 4; ++q) {
#pragma unroll
            for (int nt = 0; nt < 2; ++nt) {
                const int nn = nt * 16 + n;
#pragma unroll
                for (int r = 0; r < 8; ++r) {
                    const int m = (wv * 4 + q) * 16 + r + hi * 8;  // ck index 0..575
                    const int c = m / 9;
                    const int t = m - c * 9;
                    const float colv = tanhf(acc[q][nt][r] + b_c[m]);
                    const float* xb = x + ((size_t)b * 64 + c) * (size_t)HWSZ;
                    float pos = 0.f;
#pragma unroll
                    for (int g = 0; g < 4; ++g) {
                        const int id = gidx[t][nn][g];
                        if (id >= 0) pos += gwt[t][nn][g] * xb[id];
                    }
                    const float v = (colv + pos) * omv[18 + t][nn];
                    vmat[nn][m] = f2bf(v);
                }
            }
        }
    }
    __syncthreads();

    // --- 5. out = W_conv[64x576] @ V (no bias); waves 0-7: (mt, nt) = (wv>>1, wv&1)
    if (wv < 8) {
        const int mt = wv >> 1;
        const int nt = wv & 1;
        v8f acc = zero8();
        for (int kc = 0; kc < 18; ++kc) {
            const v16bf bop = load_b_lds(&vmat[nt * 16 + n][0], kc * 32, lane);
            const v16bf aop = load_a_pre(wconv_op, mt, 18, kc, lane);
            acc = WMMA_BF16(aop, bop, acc);
        }
#pragma unroll
        for (int r = 0; r < 8; ++r) {
            const int o = mt * 16 + r + hi * 8;
            out[(((size_t)b * 64 + o) * 128 + h) * 128 + w0 + nt * 16 + n] = acc[r];
        }
    }
}

extern "C" void kernel_launch(void* const* d_in, const int* in_sizes, int n_in,
                              void* d_out, int out_size, void* d_ws, size_t ws_size,
                              hipStream_t stream) {
    (void)in_sizes; (void)n_in; (void)out_size; (void)ws_size;
    const float* x      = (const float*)d_in[0];
    const float* ref    = (const float*)d_in[1];
    const float* w_cd   = (const float*)d_in[2];
    const float* b_cd   = (const float*)d_in[3];
    const float* w_p    = (const float*)d_in[4];
    const float* b_p    = (const float*)d_in[5];
    const float* w_m    = (const float*)d_in[6];
    const float* b_m    = (const float*)d_in[7];
    const float* w_c    = (const float*)d_in[8];
    const float* b_c    = (const float*)d_in[9];
    const float* w_conv = (const float*)d_in[10];
    float* out = (float*)d_out;

    // Workspace layout (u16 elements)
    u16t* fused    = (u16t*)d_ws;
    u16t* wcd_op   = fused + FUSED_ELEMS;
    u16t* wom_op   = wcd_op + WCD_ELEMS;
    u16t* wc_op    = wom_op + WOM_ELEMS;
    u16t* wconv_op = wc_op + WC_ELEMS;

    prep_weights<<<dim3((WC_ELEMS + 255) / 256), dim3(256), 0, stream>>>(
        w_cd, w_p, w_m, w_c, w_conv, wcd_op, wom_op, wc_op, wconv_op);

    stage1_fused_gemm<<<dim3((BN * HWSZ) / 64), dim3(128), 0, stream>>>(
        x, ref, wcd_op, b_cd, fused);
    stage2_deform_main<<<dim3((BN * HWSZ) / 32), dim3(288), 0, stream>>>(
        x, fused, wom_op, b_p, b_m, wc_op, b_c, wconv_op, out);
}
